// WaveletHead_69698729279506
// MI455X (gfx1250) — matrix operand order repaired and checked by reading, not compiled, not run on hardware
//
#include <hip/hip_runtime.h>
#include <hip/hip_bf16.h>

// ---------------------------------------------------------------------------
// WaveletHead for MI455X (gfx1250)
//   - bf16 WMMA GEMM (v_wmma_f32_16x16x32_bf16), split-K f32 atomics
//   - double-buffered LDS staging; A panel via async global->LDS (ASYNCcnt)
//   - W streamed with non-temporal loads (keep A L2-resident)
//   - ISWT (db4, periodization, 3 levels) epilogue
// ---------------------------------------------------------------------------
typedef __attribute__((ext_vector_type(16))) __bf16 v16bf;
typedef __attribute__((ext_vector_type(8)))  float  v8f;

#define BATCH    32
#define NVARS    7
#define MROWS    224                 // BATCH * NVARS
#define DIN      49152               // D_MODEL * PATCH_NUMS
#define PRED     336
#define BANDS    4
#define LEVEL    3

#define KSPLIT   4
#define KSEG     (DIN / KSPLIT)      // 12288
#define KT       64                  // K chunk staged in LDS
#define NCHUNK   (KSEG / KT)         // 192
#define NSTRIP   48                  // output columns per workgroup (3 x 16)
#define GEMM_THREADS 448             // 14 waves: one 16-row M tile per wave
#define ASTR     72                  // LDS row stride (bf16 elems), padded
#define BSTR     72
#define BELEMS   (KT * NSTRIP)       // 3072 -> 7 per-thread units

// db4 reconstruction filters (pywt convention)
__device__ __constant__ float c_rec_lo[8] = {
    0.23037781330885523f,  0.7148465705525415f,   0.6308807679295904f,
   -0.02798376941698385f, -0.18703481171888114f,  0.030841381835986965f,
    0.032883011666982945f,-0.010597401784997278f };
__device__ __constant__ float c_rec_hi[8] = {
    0.010597401784997278f, 0.032883011666982945f,-0.030841381835986965f,
   -0.18703481171888114f,  0.02798376941698385f,  0.6308807679295904f,
   -0.7148465705525415f,   0.23037781330885523f };

__device__ __forceinline__ unsigned short f2bf(float f) {
    union { float f; unsigned u; } v; v.f = f;
    unsigned r = (v.u + 0x7FFFu + ((v.u >> 16) & 1u)) >> 16;   // RNE
    return (unsigned short)r;
}

__device__ __forceinline__ void wait_async0() {
#if __has_builtin(__builtin_amdgcn_s_wait_asynccnt)
    __builtin_amdgcn_s_wait_asynccnt(0);
#else
    asm volatile("s_wait_asynccnt 0x0" ::: "memory");
#endif
}

// ---------------------------------------------------------------------------
// Kernel 1: convert x (fp32, 224 x 49152) -> bf16 panel in workspace
// ---------------------------------------------------------------------------
__global__ void wh_cvt_a(const float* __restrict__ x, unsigned short* __restrict__ abf,
                         long long total) {
    long long i = ((long long)blockIdx.x * blockDim.x + threadIdx.x) * 4;
    if (i + 3 >= total) {
        for (long long j = i; j < total; ++j) abf[j] = f2bf(x[j]);
        return;
    }
    float4 f = *(const float4*)(x + i);
    unsigned lo = (unsigned)f2bf(f.x) | ((unsigned)f2bf(f.y) << 16);
    unsigned hi = (unsigned)f2bf(f.z) | ((unsigned)f2bf(f.w) << 16);
    uint2 p; p.x = lo; p.y = hi;
    *(uint2*)(abf + i) = p;
}

// ---------------------------------------------------------------------------
// Kernel 2: coeffs[band][row][p] = bias[band][p]
// ---------------------------------------------------------------------------
__global__ void wh_init_coeffs(float* __restrict__ coeffs, const float* __restrict__ bias) {
    int idx = blockIdx.x * blockDim.x + threadIdx.x;
    if (idx >= BANDS * MROWS * PRED) return;
    int k = idx / (MROWS * PRED);
    int p = idx % PRED;
    coeffs[idx] = bias[k * PRED + p];
}

// ---------------------------------------------------------------------------
// GEMM helpers
// ---------------------------------------------------------------------------
// Issue async global->LDS copies of the bf16 A chunk (224 x 64), 16B per op.
__device__ __forceinline__ void stage_a_async(const unsigned short* __restrict__ abf,
                                              unsigned short* buf, int kd0, int tid) {
    #pragma unroll
    for (int it = 0; it < 4; ++it) {
        int u  = it * GEMM_THREADS + tid;      // 0..1791
        int r  = u >> 3;                       // row 0..223
        int c8 = (u & 7) * 8;                  // k sub-offset 0..56
        unsigned lds_addr =
            (unsigned)(size_t)(buf + r * ASTR + c8);           // low 32 = LDS offset
        unsigned long long ga =
            (unsigned long long)(size_t)(abf + (size_t)r * DIN + kd0 + c8);
        asm volatile("global_load_async_to_lds_b128 %0, %1, off"
                     :: "v"(lds_addr), "v"(ga) : "memory");
    }
}

// Issue the fp32 W chunk loads (non-temporal: W streams exactly once).
__device__ __forceinline__ void b_issue(const float* __restrict__ Wk, int kd0, int p0,
                                        int tid, float* wv) {
    #pragma unroll
    for (int i = 0; i < 7; ++i) {
        int u = i * GEMM_THREADS + tid;
        if (u < BELEMS) {
            int d = u / NSTRIP, p = u % NSTRIP;
            wv[i] = __builtin_nontemporal_load(Wk + (size_t)(kd0 + d) * PRED + p0 + p);
        }
    }
}

// Convert held W values and scatter transposed [col][k] into LDS.
__device__ __forceinline__ void b_store(unsigned short* buf, int tid, const float* wv) {
    #pragma unroll
    for (int i = 0; i < 7; ++i) {
        int u = i * GEMM_THREADS + tid;
        if (u < BELEMS) {
            int d = u / NSTRIP, p = u % NSTRIP;
            buf[p * BSTR + d] = f2bf(wv[i]);
        }
    }
}

// Two K=32 WMMA steps over one staged chunk (3 output column tiles per wave).
__device__ __forceinline__ void do_compute(const unsigned short* Ab, const unsigned short* Bb,
                                           int mr, int l, int hf, v8f* acc) {
    #pragma unroll
    for (int kk = 0; kk < KT; kk += 32) {
        union { v16bf v; unsigned u[8]; } af;
        #pragma unroll
        for (int v = 0; v < 8; ++v) {
            // ISA 16-bit A layout: half 0 -> K {0..7,16..23}, half 1 -> +8
            int ks = (v < 4 ? 2 * v : 2 * v + 8) + 8 * hf + kk;
            af.u[v] = *(const unsigned*)&Ab[(mr + l) * ASTR + ks];
        }
        #pragma unroll
        for (int j = 0; j < 3; ++j) {
            union { v16bf v; unsigned u[8]; } bv;
            #pragma unroll
            for (int v = 0; v < 8; ++v) {
                // same K order as A: half 0 -> K {0..7,16..23}, half 1 -> +8
                int ks = (v < 4 ? 2 * v : 2 * v + 8) + 8 * hf + kk;
                bv.u[v] = *(const unsigned*)&Bb[(j * 16 + l) * BSTR + ks];
            }
            acc[j] = __builtin_amdgcn_wmma_f32_16x16x32_bf16(
                false, af.v, false, bv.v, (short)0, acc[j], false, false);
        }
    }
}

// ---------------------------------------------------------------------------
// Kernel 3: GEMM  coeffs[band][r][p] += sum_d A[r][d] * W[band][d][p]
//   grid = (7 col-strips, 4 bands, KSPLIT); 448 thr = 14 waves
//   double-buffered LDS, async A staging, prefetch-under-WMMA for W
// ---------------------------------------------------------------------------
__global__ __launch_bounds__(GEMM_THREADS)
void wh_gemm(const unsigned short* __restrict__ abf,   // bf16 A, 224 x 49152
             const float* __restrict__ W,              // fp32, [band][49152][336]
             float* __restrict__ coeffs) {             // f32,  [band][224][336]
    __shared__ unsigned short A_lds[2][MROWS * ASTR];  // 2 x 31.5 KB
    __shared__ unsigned short B_lds[2][NSTRIP * BSTR]; // 2 x 6.75 KB

    const int tid  = threadIdx.x;
    const int wave = tid >> 5;
    const int lane = tid & 31;
    const int l    = lane & 15;
    const int hf   = (lane >> 4) & 1;

    const int mr   = wave * 16;
    const int p0   = blockIdx.x * NSTRIP;
    const int band = blockIdx.y;
    const int kd0b = blockIdx.z * KSEG;

    const float* Wk = W + (size_t)band * DIN * PRED;

    v8f acc[3] = {};

    // ---- prologue: stage chunk 0 into buffer 0 ----
    {
        float wv[7];
        b_issue(Wk, kd0b, p0, tid, wv);
        stage_a_async(abf, A_lds[0], kd0b, tid);
        b_store(B_lds[0], tid, wv);
        wait_async0();
        __syncthreads();
    }

    for (int c = 0; c < NCHUNK; ++c) {
        const int cur = c & 1;
        const int nxt = cur ^ 1;
        const bool pf = (c + 1 < NCHUNK);
        float wv[7];
        if (pf) {
            // issue next chunk's traffic before the math so it hides under WMMA
            b_issue(Wk, kd0b + (c + 1) * KT, p0, tid, wv);
            stage_a_async(abf, A_lds[nxt], kd0b + (c + 1) * KT, tid);
        }

        do_compute(A_lds[cur], B_lds[cur], mr, l, hf, acc);

        if (pf) b_store(B_lds[nxt], tid, wv);   // loadcnt wait lands here, post-WMMA
        wait_async0();
        __syncthreads();
    }

    // ---- split-K accumulate: D layout rows = r + 8*half, cols = lane%16 ----
    #pragma unroll
    for (int j = 0; j < 3; ++j) {
        #pragma unroll
        for (int r = 0; r < 8; ++r) {
            int rowg = mr + r + 8 * hf;
            int colg = p0 + j * 16 + l;
            atomicAdd(&coeffs[((size_t)band * MROWS + rowg) * PRED + colg], acc[j][r]);
        }
    }
}

// ---------------------------------------------------------------------------
// Kernel 4: inverse SWT (db4, periodization, 3 levels), one block per row
// ---------------------------------------------------------------------------
__device__ __forceinline__ float idwt_y(const float* ca, const float* cd,
                                        int t, int par, int s, int step, int M) {
    float acc = 0.f;
    #pragma unroll
    for (int k = 0; k < 8; ++k) {
        int j2 = t - k; if (j2 < 0) j2 += M;        // (t-k) mod M, M even
        if ((j2 & 1) == 0) {                        // upsampled: odd slots are zero
            int a = ((j2 >> 1) * 2 + par) * step + s;
            acc += ca[a] * c_rec_lo[k] + cd[a] * c_rec_hi[k];
        }
    }
    return acc;
}

__global__ void wh_iswt(const float* __restrict__ coeffs, float* __restrict__ out) {
    const int row = blockIdx.x;
    const int tid = threadIdx.x;
    __shared__ float cur[PRED], nxt[PRED], cdb[PRED];

    for (int t = tid; t < PRED; t += blockDim.x)
        cur[t] = coeffs[(size_t)row * PRED + t];                 // band 0 = cA
    __syncthreads();

    for (int j = LEVEL; j >= 1; --j) {
        const int step = 1 << (j - 1);
        const int M    = PRED / step;
        const int bi   = LEVEL - j + 1;                          // cD band index
        for (int t = tid; t < PRED; t += blockDim.x)
            cdb[t] = coeffs[((size_t)bi * MROWS + row) * PRED + t];
        __syncthreads();

        for (int idx = tid; idx < PRED; idx += blockDim.x) {
            int s = idx % step, m = idx / step;
            int te = m + 3; if (te >= M) te -= M;                // roll(y,-3): even branch
            int to = m + 2; if (to >= M) to -= M;                // odd branch + roll(.,1)
            float ye = idwt_y(cur, cdb, te, 0, s, step, M);
            float yo = idwt_y(cur, cdb, to, 1, s, step, M);
            nxt[idx] = 0.5f * (ye + yo);
        }
        __syncthreads();
        for (int t = tid; t < PRED; t += blockDim.x) cur[t] = nxt[t];
        __syncthreads();
    }

    for (int t = tid; t < PRED; t += blockDim.x)
        out[(size_t)row * PRED + t] = cur[t];
}

// ---------------------------------------------------------------------------
extern "C" void kernel_launch(void* const* d_in, const int* in_sizes, int n_in,
                              void* d_out, int out_size, void* d_ws, size_t ws_size,
                              hipStream_t stream) {
    const float* x    = (const float*)d_in[0];   // (32,7,768,64)
    const float* W    = (const float*)d_in[1];   // (4,49152,336)
    const float* bias = (const float*)d_in[2];   // (4,336)
    float*       out  = (float*)d_out;           // (32,7,336)

    // workspace: bf16 A panel (22.0 MB) then f32 coeffs (1.2 MB)
    unsigned short* abf = (unsigned short*)d_ws;
    const size_t abf_bytes = (size_t)MROWS * DIN * sizeof(unsigned short); // 22020096
    float* coeffs = (float*)((char*)d_ws + abf_bytes);

    // 1) fp32 -> bf16 A panel
    {
        long long total = (long long)MROWS * DIN;        // 11,010,048
        int thr = 256;
        int blocks = (int)((total / 4 + thr - 1) / thr); // 10752
        wh_cvt_a<<<blocks, thr, 0, stream>>>(x, abf, total);
    }
    // 2) coeffs = broadcast bias
    {
        int total = BANDS * MROWS * PRED;                // 301,056
        wh_init_coeffs<<<(total + 255) / 256, 256, 0, stream>>>(coeffs, bias);
    }
    // 3) bf16 WMMA GEMM with split-K atomic accumulation
    {
        dim3 grid(PRED / NSTRIP, BANDS, KSPLIT);         // (7,4,4)
        wh_gemm<<<grid, GEMM_THREADS, 0, stream>>>(abf, W, coeffs);
    }
    // 4) inverse SWT epilogue
    {
        wh_iswt<<<MROWS, 128, 0, stream>>>(coeffs, out);
    }
}
